// TorchS5_8117488189492
// MI455X (gfx1250) — matrix operand order repaired and checked by reading, not compiled, not run on hardware
//
#include <hip/hip_runtime.h>
#include <hip/hip_bf16.h>
#include <stddef.h>

// ---------------------------------------------------------------------------
// S5 SSM layer for MI455X (gfx1250, wave32, WMMA).
//   GEMM1 (u @ B^T) + ZOH discretization  -> A_bar, b_seq   [bf16 WMMA]
//   sequential scan over L                -> x (in place)    [f32 VALU]
//   GEMM2 (x @ C^T) + D*u                 -> ys              [bf16 WMMA]
// Double-buffered LDS staging; global loads for tile k+1 issued before the
// WMMAs of tile k so VMEM latency hides under matrix work.
// ---------------------------------------------------------------------------

typedef __attribute__((ext_vector_type(16))) __bf16 v16bf;
typedef __attribute__((ext_vector_type(8)))  float  v8f;

constexpr int H_IN   = 1024;
constexpr int H_OUT  = 1024;
constexpr int PDIM   = 512;
constexpr int BSZ    = 16;
constexpr int LSEQ   = 1024;
constexpr int MROWS  = BSZ * LSEQ;   // 16384

constexpr int TILE_M = 128;          // block tile rows (tokens)
constexpr int TILE_N = 64;           // block tile cols
constexpr int KB     = 32;           // WMMA K per step (bf16)

union Frag16 { v16bf v; uint4 q[2]; };      // 16 bf16 = 32 B = 8 VGPRs
union PackB4 { __bf16 h[4]; uint2 u2; };    // 4 bf16 = 8 B (one ds_store_b64)

// ---------------------------------------------------------------------------
// prep: weights -> bf16 ; Lam = -(softplus(L_raw)+1e-3) ; step = exp(log_step)
// ---------------------------------------------------------------------------
__global__ __launch_bounds__(256) void prep_kernel(
    const float* __restrict__ Braw, const float* __restrict__ Craw,
    const float* __restrict__ Lraw, const float* __restrict__ lstep,
    __bf16* __restrict__ Bw, __bf16* __restrict__ Cw,
    float* __restrict__ Lam, float* __restrict__ Stepv)
{
  const int i = blockIdx.x * 256 + threadIdx.x;
  if (i < PDIM * H_IN)  Bw[i] = (__bf16)Braw[i];
  if (i < H_OUT * PDIM) Cw[i] = (__bf16)Craw[i];
  if (i < PDIM) {
    const float x  = Lraw[i];
    const float sp = (x > 20.f) ? x : log1pf(__expf(x));   // softplus
    Lam[i]   = -(sp + 1e-3f);                              // MIN_NEG
    Stepv[i] = __expf(lstep[i]);                           // STEP_RESCALE=1
  }
}

// ---------------------------------------------------------------------------
// GEMM1: Bu[m,p] = sum_h u[m,h]*B[p,h], fused ZOH discretization epilogue.
// 256 threads = 8 waves (4x2 over the 128x64 tile), 32x32 subtile per wave.
// ---------------------------------------------------------------------------
__global__ __launch_bounds__(256) void gemm1_disc_kernel(
    const float*  __restrict__ u,  const __bf16* __restrict__ Bw,
    const float*  __restrict__ ts, const float*  __restrict__ Lam,
    const float*  __restrict__ Stepv,
    float* __restrict__ Abar, float* __restrict__ Bseq)
{
  __shared__ __align__(16) __bf16 sA[2][TILE_M][KB];   // 16 KB
  __shared__ __align__(16) __bf16 sB[2][TILE_N][KB];   //  8 KB

  const int tid  = threadIdx.x;
  const int lane = tid & 31;
  const int wave = tid >> 5;
  const int m0 = blockIdx.y * TILE_M;
  const int n0 = blockIdx.x * TILE_N;
  const int wm = (wave & 3) * 32;   // wave position in tile (M)
  const int wn = (wave >> 2) * 32;  // wave position in tile (N)

  // staging coordinates (uniform trip counts, fully unrolled)
  const int rA = tid >> 3, cA = (tid & 7) * 4;   // rows rA, rA+32, rA+64, rA+96
  const int rB = tid >> 2, cB = (tid & 3) * 8;

  v8f acc[2][2] = {};
  float4 ra[4];
  uint4  rb;

  // preload K-tile 0 into registers (4 independent b128 loads + 1 for B)
#pragma unroll
  for (int it = 0; it < 4; ++it)
    ra[it] = *(const float4*)(u + (size_t)(m0 + rA + 32 * it) * H_IN + cA);
  rb = *(const uint4*)(Bw + (size_t)(n0 + rB) * H_IN + cB);

  int buf = 0;
  for (int k0 = 0; k0 < H_IN; k0 += KB) {
    // commit staged registers to LDS[buf] (cvt_pk_bf16 + b64 stores)
#pragma unroll
    for (int it = 0; it < 4; ++it) {
      PackB4 pk;
      pk.h[0] = (__bf16)ra[it].x; pk.h[1] = (__bf16)ra[it].y;
      pk.h[2] = (__bf16)ra[it].z; pk.h[3] = (__bf16)ra[it].w;
      *(uint2*)&sA[buf][rA + 32 * it][cA] = pk.u2;
    }
    *(uint4*)&sB[buf][rB][cB] = rb;
    __syncthreads();

    // issue global loads for the NEXT K-tile (latency hidden under WMMAs)
    const int kn = k0 + KB;
    if (kn < H_IN) {
#pragma unroll
      for (int it = 0; it < 4; ++it)
        ra[it] = *(const float4*)(u + (size_t)(m0 + rA + 32 * it) * H_IN + kn + cA);
      rb = *(const uint4*)(Bw + (size_t)(n0 + rB) * H_IN + kn + cB);
    }

    // fragments per CDNA5 16-bit A/B VGPR layouts
    const int rsel  = lane & 15;
    const int khalf = (lane >> 4) * 8;
    const int kb16  = (lane >> 4) * 16;
    Frag16 a[2], b[2];
#pragma unroll
    for (int i = 0; i < 2; ++i) {
      const int row = wm + 16 * i + rsel;
      a[i].q[0] = *(const uint4*)&sA[buf][row][khalf];
      a[i].q[1] = *(const uint4*)&sA[buf][row][16 + khalf];
      const int coln = wn + 16 * i + rsel;
      b[i].q[0] = *(const uint4*)&sB[buf][coln][khalf];
      b[i].q[1] = *(const uint4*)&sB[buf][coln][16 + khalf];
    }
#pragma unroll
    for (int i = 0; i < 2; ++i)
#pragma unroll
      for (int j = 0; j < 2; ++j)
        acc[i][j] = __builtin_amdgcn_wmma_f32_16x16x32_bf16(
            false, a[i].v, false, b[j].v, (short)0, acc[i][j], false, false);
    buf ^= 1;
  }

  // Epilogue: C/D layout -> lane L, VGPR r maps to (row = r + 8*(L>=16), col = L&15)
  const int col   = lane & 15;
  const int rbase = (lane >> 4) * 8;
#pragma unroll
  for (int j = 0; j < 2; ++j) {
    const int p = n0 + wn + 16 * j + col;
    const float lam = Lam[p];
    const float st  = Stepv[p];
#pragma unroll
    for (int i = 0; i < 2; ++i) {
#pragma unroll
      for (int r = 0; r < 8; ++r) {
        const int m = m0 + wm + 16 * i + rbase + r;
        const float t     = ts[m];
        const float delta = fmaxf(st * t, 0.f);
        const float ab    = __expf(fminf(lam * delta, 20.f));   // CLIP_MAX
        const float gam   = (fabsf(lam) > 1e-6f) ? (ab - 1.f) / lam : delta;
        const size_t o = (size_t)m * PDIM + p;
        Abar[o] = ab;
        Bseq[o] = gam * acc[i][j][r];
      }
    }
  }
}

// ---------------------------------------------------------------------------
// Scan: per (b,p) chain over L. x = A_cum * cumsum(b_seq/(A_cum+eps)).
// Unrolled x8: 16 independent loads issue ahead of the dependent chain.
// ---------------------------------------------------------------------------
__global__ __launch_bounds__(256) void scan_kernel(
    const float* __restrict__ Abar, float* __restrict__ X)
{
  const int idx = blockIdx.x * 256 + threadIdx.x;   // 0 .. BSZ*PDIM-1
  const int b = idx / PDIM, p = idx % PDIM;
  size_t o = (size_t)b * LSEQ * PDIM + p;
  float ac = 1.f, s = 0.f;
  for (int l = 0; l < LSEQ; l += 8) {
    float a[8], q[8];
#pragma unroll
    for (int t = 0; t < 8; ++t) a[t] = Abar[o + (size_t)t * PDIM];
#pragma unroll
    for (int t = 0; t < 8; ++t) q[t] = X[o + (size_t)t * PDIM];
#pragma unroll
    for (int t = 0; t < 8; ++t) {
      ac *= a[t];
      s += q[t] / (ac + 1e-9f);
      X[o + (size_t)t * PDIM] = ac * s;
    }
    o += 8 * PDIM;
  }
}

// ---------------------------------------------------------------------------
// GEMM2: ys[m,h] = sum_p C[h,p]*x[m,p] + D[h]*u[m,h]
// ---------------------------------------------------------------------------
__global__ __launch_bounds__(256) void gemm2_out_kernel(
    const float* __restrict__ X,  const __bf16* __restrict__ Cw,
    const float* __restrict__ Dv, const float*  __restrict__ u,
    float* __restrict__ ys)
{
  __shared__ __align__(16) __bf16 sA[2][TILE_M][KB];
  __shared__ __align__(16) __bf16 sB[2][TILE_N][KB];

  const int tid  = threadIdx.x;
  const int lane = tid & 31;
  const int wave = tid >> 5;
  const int m0 = blockIdx.y * TILE_M;
  const int n0 = blockIdx.x * TILE_N;
  const int wm = (wave & 3) * 32;
  const int wn = (wave >> 2) * 32;

  const int rA = tid >> 3, cA = (tid & 7) * 4;
  const int rB = tid >> 2, cB = (tid & 3) * 8;

  v8f acc[2][2] = {};
  float4 ra[4];
  uint4  rb;

#pragma unroll
  for (int it = 0; it < 4; ++it)
    ra[it] = *(const float4*)(X + (size_t)(m0 + rA + 32 * it) * PDIM + cA);
  rb = *(const uint4*)(Cw + (size_t)(n0 + rB) * PDIM + cB);

  int buf = 0;
  for (int k0 = 0; k0 < PDIM; k0 += KB) {
#pragma unroll
    for (int it = 0; it < 4; ++it) {
      PackB4 pk;
      pk.h[0] = (__bf16)ra[it].x; pk.h[1] = (__bf16)ra[it].y;
      pk.h[2] = (__bf16)ra[it].z; pk.h[3] = (__bf16)ra[it].w;
      *(uint2*)&sA[buf][rA + 32 * it][cA] = pk.u2;
    }
    *(uint4*)&sB[buf][rB][cB] = rb;
    __syncthreads();

    const int kn = k0 + KB;
    if (kn < PDIM) {
#pragma unroll
      for (int it = 0; it < 4; ++it)
        ra[it] = *(const float4*)(X + (size_t)(m0 + rA + 32 * it) * PDIM + kn + cA);
      rb = *(const uint4*)(Cw + (size_t)(n0 + rB) * PDIM + kn + cB);
    }

    const int rsel  = lane & 15;
    const int khalf = (lane >> 4) * 8;
    const int kb16  = (lane >> 4) * 16;
    Frag16 a[2], b[2];
#pragma unroll
    for (int i = 0; i < 2; ++i) {
      const int row = wm + 16 * i + rsel;
      a[i].q[0] = *(const uint4*)&sA[buf][row][khalf];
      a[i].q[1] = *(const uint4*)&sA[buf][row][16 + khalf];
      const int coln = wn + 16 * i + rsel;
      b[i].q[0] = *(const uint4*)&sB[buf][coln][khalf];
      b[i].q[1] = *(const uint4*)&sB[buf][coln][16 + khalf];
    }
#pragma unroll
    for (int i = 0; i < 2; ++i)
#pragma unroll
      for (int j = 0; j < 2; ++j)
        acc[i][j] = __builtin_amdgcn_wmma_f32_16x16x32_bf16(
            false, a[i].v, false, b[j].v, (short)0, acc[i][j], false, false);
    buf ^= 1;
  }

  const int col   = lane & 15;
  const int rbase = (lane >> 4) * 8;
#pragma unroll
  for (int j = 0; j < 2; ++j) {
    const int h  = n0 + wn + 16 * j + col;
    const float dj = Dv[h];
#pragma unroll
    for (int i = 0; i < 2; ++i) {
#pragma unroll
      for (int r = 0; r < 8; ++r) {
        const int m = m0 + wm + 16 * i + rbase + r;
        ys[(size_t)m * H_OUT + h] =
            acc[i][j][r] + dj * u[(size_t)m * H_IN + h];
      }
    }
  }
}

// ---------------------------------------------------------------------------
extern "C" void kernel_launch(void* const* d_in, const int* in_sizes, int n_in,
                              void* d_out, int out_size, void* d_ws, size_t ws_size,
                              hipStream_t stream) {
  (void)in_sizes; (void)n_in; (void)out_size; (void)ws_size;
  const float* u     = (const float*)d_in[0];   // [16,1024,1024]
  const float* ts    = (const float*)d_in[1];   // [16,1024]
  const float* Lraw  = (const float*)d_in[2];   // [512]
  const float* lstep = (const float*)d_in[3];   // [512]
  const float* Braw  = (const float*)d_in[4];   // [512,1024]
  const float* Craw  = (const float*)d_in[5];   // [1024,512]
  const float* Dv    = (const float*)d_in[6];   // [1024]
  float* ys = (float*)d_out;

  // workspace layout (~69.2 MB total)
  char* ws = (char*)d_ws;
  __bf16* Bw    = (__bf16*)ws;                          // 1 MB
  __bf16* Cw    = Bw + (size_t)PDIM * H_IN;             // 1 MB
  float*  Lam   = (float*)(Cw + (size_t)H_OUT * PDIM);  // 2 KB
  float*  Stepv = Lam + PDIM;                           // 2 KB
  float*  Abar  = Stepv + PDIM;                         // 33.6 MB
  float*  Bseq  = Abar + (size_t)MROWS * PDIM;          // 33.6 MB (becomes x)

  prep_kernel<<<(PDIM * H_IN + 255) / 256, 256, 0, stream>>>(
      Braw, Craw, Lraw, lstep, Bw, Cw, Lam, Stepv);

  dim3 g1(PDIM / TILE_N, MROWS / TILE_M);    // (8, 128)
  gemm1_disc_kernel<<<g1, 256, 0, stream>>>(u, Bw, ts, Lam, Stepv, Abar, Bseq);

  scan_kernel<<<(BSZ * PDIM) / 256, 256, 0, stream>>>(Abar, Bseq);

  dim3 g2(H_OUT / TILE_N, MROWS / TILE_M);   // (16, 128)
  gemm2_out_kernel<<<g2, 256, 0, stream>>>(Bseq, Cw, Dv, u, ys);
}
